// BiSRU_8443905704271
// MI455X (gfx1250) — matrix operand order repaired and checked by reading, not compiled, not run on hardware
//
#include <hip/hip_runtime.h>
#include <hip/hip_bf16.h>

// Problem dims (fixed by the reference)
#define TT 512
#define BB 64
#define VV 50000
#define DD 300
#define DP 320        // D padded to multiple of 32 for bf16 WMMA K
#define HH 512
#define CC 10
#define TB (TT*BB)    // 32768 GEMM rows
#define NN (4*HH)     // 2048 GEMM cols
#define K1 (2*HH)     // 1024 layer-1 K

// Flip to 0 if the gfx1250 async mnemonics are rejected by the assembler.
#define USE_ASYNC_LDS 1

typedef unsigned short ushort_t;
typedef __attribute__((ext_vector_type(16))) __bf16 v16bf;
typedef __attribute__((ext_vector_type(8)))  float  v8f;

__device__ __forceinline__ ushort_t f2bf(float f) {
    unsigned u = __float_as_uint(f);
    unsigned r = (u + 0x7FFFu + ((u >> 16) & 1u)) >> 16;   // round-to-nearest-even
    return (ushort_t)r;
}

// --- CDNA5 async global->LDS copy (ASYNCcnt tracked, no VGPR bounce) -------
__device__ __forceinline__ void async_ld_b128(const ushort_t* g, ushort_t* lds) {
#if USE_ASYNC_LDS
    unsigned lds_off = (unsigned)(unsigned long long)lds;  // low 32 bits = LDS offset
    asm volatile("global_load_async_to_lds_b128 %0, %1, off"
                 :: "v"(lds_off), "v"(g) : "memory");
#else
    *(uint4*)lds = *(const uint4*)g;
#endif
}
__device__ __forceinline__ void wait_asynccnt0() {
#if USE_ASYNC_LDS
    asm volatile("s_wait_asynccnt 0" ::: "memory");
#endif
}

// ---------------------------------------------------------------- embedding
__global__ void embed_gather_kernel(const int* __restrict__ x,
                                    const float* __restrict__ embed,
                                    ushort_t* __restrict__ e_bf) {
    int tb  = blockIdx.x;                 // 0..TB-1 (row = t*B + b)
    int row = x[tb];
    const float* src = embed + (size_t)row * DD;
    ushort_t*    dst = e_bf + (size_t)tb * DP;
    for (int c = threadIdx.x; c < DP; c += blockDim.x)
        dst[c] = (c < DD) ? f2bf(src[c]) : (ushort_t)0;
}

// ------------------------------------- weights f32 [K,N] -> bf16 [N,Kp] (T)
// Transposed so the GEMM B tile is a plain row-major copy (async-friendly).
__global__ void cvt_weight_t_kernel(const float* __restrict__ src,
                                    ushort_t* __restrict__ dst,
                                    int K, int Kp, int N) {
    int i = blockIdx.x * blockDim.x + threadIdx.x;   // over N*Kp
    if (i >= N * Kp) return;
    int n = i / Kp, k = i - n * Kp;
    dst[i] = (k < K) ? f2bf(src[(size_t)k * N + n]) : (ushort_t)0;
}

// ---------------------------------------------------------------- WMMA GEMM
// C[M,N] f32 = A[M,K] bf16 @ Bt[N,K] bf16 (B pre-transposed).
// M%128==0, N%128==0, K%32==0.
#define MBLK 128
#define NBLK 128
#define KSTEP 32
#define LSTR  40   // padded LDS K-stride: 20*row mod 64 -> conflict-free b128 reads

__global__ __launch_bounds__(256)
void wmma_gemm_bf16_kernel(const ushort_t* __restrict__ A,
                           const ushort_t* __restrict__ Bt,
                           float* __restrict__ C,
                           int M, int N, int K) {
    __shared__ __align__(16) ushort_t Alds[2][MBLK * LSTR];   // [row][k]
    __shared__ __align__(16) ushort_t Blds[2][NBLK * LSTR];   // [n][k]

    const int tid   = threadIdx.x;
    const int m_blk = blockIdx.y * MBLK;
    const int n_blk = blockIdx.x * NBLK;
    const int wave  = tid >> 5;
    const int lane  = tid & 31;
    const int wm    = wave & 1;    // 2 wave rows  (64 M each)
    const int wn    = wave >> 1;   // 4 wave cols  (32 N each)
    const int half  = lane >> 4;   // lane half selects K sub-range
    const int lm    = lane & 15;

    // Per-thread staging coordinates: 2 passes x (1 A chunk + 1 B chunk) of 16B
    const int srow0 = tid >> 2, sc8_0 = (tid & 3) * 8;          // idx 0..255
    const int srow1 = (tid + 256) >> 2, sc8_1 = sc8_0;          // idx 256..511

    v8f acc[4][2];
    #pragma unroll
    for (int i = 0; i < 4; ++i)
        #pragma unroll
        for (int j = 0; j < 2; ++j) { v8f z = {}; acc[i][j] = z; }

    const int nk = K / KSTEP;

    // ---- stage tile 0 into buffer 0
    {
        const int k0 = 0;
        async_ld_b128(A  + (size_t)(m_blk + srow0) * K + k0 + sc8_0, &Alds[0][srow0 * LSTR + sc8_0]);
        async_ld_b128(A  + (size_t)(m_blk + srow1) * K + k0 + sc8_1, &Alds[0][srow1 * LSTR + sc8_1]);
        async_ld_b128(Bt + (size_t)(n_blk + srow0) * K + k0 + sc8_0, &Blds[0][srow0 * LSTR + sc8_0]);
        async_ld_b128(Bt + (size_t)(n_blk + srow1) * K + k0 + sc8_1, &Blds[0][srow1 * LSTR + sc8_1]);
    }
    wait_asynccnt0();
    __syncthreads();

    for (int i = 0; i < nk; ++i) {
        const int cur = i & 1;
        // ---- issue async stage of next tile into the other buffer
        if (i + 1 < nk) {
            const int k0 = (i + 1) * KSTEP;
            const int nxt = cur ^ 1;
            async_ld_b128(A  + (size_t)(m_blk + srow0) * K + k0 + sc8_0, &Alds[nxt][srow0 * LSTR + sc8_0]);
            async_ld_b128(A  + (size_t)(m_blk + srow1) * K + k0 + sc8_1, &Alds[nxt][srow1 * LSTR + sc8_1]);
            async_ld_b128(Bt + (size_t)(n_blk + srow0) * K + k0 + sc8_0, &Blds[nxt][srow0 * LSTR + sc8_0]);
            async_ld_b128(Bt + (size_t)(n_blk + srow1) * K + k0 + sc8_1, &Blds[nxt][srow1 * LSTR + sc8_1]);
        }

        // ---- assemble fragments per ISA 16-bit layouts and multiply
        union Frag { v16bf v; uint4 q[2]; } af[4], bf[2];
        const int kbA = half * 8;    // A: lanes 0-15 K0..7/16..23, lanes16-31 +8
        const int kbB = half * 8;    // B: lanes 0-15 K0..7/16..23, lanes16-31 +8
        #pragma unroll
        for (int ii = 0; ii < 4; ++ii) {
            int row = wm * 64 + ii * 16 + lm;
            af[ii].q[0] = *(const uint4*)&Alds[cur][row * LSTR + kbA];
            af[ii].q[1] = *(const uint4*)&Alds[cur][row * LSTR + kbA + 16];
        }
        #pragma unroll
        for (int jj = 0; jj < 2; ++jj) {
            int col = wn * 32 + jj * 16 + lm;
            bf[jj].q[0] = *(const uint4*)&Blds[cur][col * LSTR + kbB];
            bf[jj].q[1] = *(const uint4*)&Blds[cur][col * LSTR + kbB + 16];
        }
        #pragma unroll
        for (int ii = 0; ii < 4; ++ii)
            #pragma unroll
            for (int jj = 0; jj < 2; ++jj)
                acc[ii][jj] = __builtin_amdgcn_wmma_f32_16x16x32_bf16(
                    false, af[ii].v, false, bf[jj].v, (short)0, acc[ii][jj], false, false);

        if (i + 1 < nk) wait_asynccnt0();   // own async writes of next tile done
        __syncthreads();                    // everyone's writes visible / reads done
    }

    // ---- store D: lane = N, vgpr r = M (lanes 16-31 -> M+8)
    #pragma unroll
    for (int i = 0; i < 4; ++i)
        #pragma unroll
        for (int j = 0; j < 2; ++j) {
            int col = n_blk + wn * 32 + j * 16 + lm;
            #pragma unroll
            for (int r = 0; r < 8; ++r) {
                int row = m_blk + wm * 64 + i * 16 + r + half * 8;
                C[(size_t)row * N + col] = acc[i][j][r];
            }
        }
}

// ------------------------------------------------- bidirectional SRU scan
// One thread per (dir, b, h).  dir 0: forward from Uf, cols [0,H).
//                              dir 1: reverse from Ub, cols [H,2H).
// Layer 0 writes bf16 activations; layer 1 computes max_t tanh(h) (pooled).
__global__ void sru_scan_bidir_kernel(const float* __restrict__ Uf,
                                      const float* __restrict__ Ub,
                                      const float* __restrict__ bias_f,
                                      const float* __restrict__ bias_b,
                                      ushort_t* __restrict__ out_bf,   // or null
                                      float* __restrict__ out_pooled)  // or null
{
    int tid = blockIdx.x * blockDim.x + threadIdx.x;   // 0..2*B*H-1
    int dir = tid >> 15;
    int r   = tid & 32767;
    int b   = r >> 9;
    int h   = r & (HH - 1);

    const float* U    = dir ? Ub : Uf;
    const float* bias = dir ? bias_b : bias_f;
    const float  bfv  = bias[h];
    const float  brv  = bias[HH + h];

    float c = 0.0f, m = -1.0f;
    for (int s = 0; s < TT; ++s) {
        int t = dir ? (TT - 1 - s) : s;
        size_t base = ((size_t)t * BB + b) * NN;
        float xt = U[base + h];
        float fz = U[base + HH + h];
        float rz = U[base + 2 * HH + h];
        float hw = U[base + 3 * HH + h];
        float f  = 1.0f / (1.0f + __expf(-(fz + bfv)));
        float rr = 1.0f / (1.0f + __expf(-(rz + brv)));
        c = f * c + (1.0f - f) * xt;
        float o = rr * tanhf(c) + (1.0f - rr) * hw;
        if (out_bf)
            out_bf[((size_t)t * BB + b) * K1 + dir * HH + h] = f2bf(o);
        else
            m = fmaxf(m, tanhf(o));
    }
    if (out_pooled)
        out_pooled[(size_t)b * K1 + dir * HH + h] = m;
}

// ---------------------------------------------------------------- head
__global__ void head_kernel(const float* __restrict__ pooled,
                            const float* __restrict__ Wh,
                            const float* __restrict__ bh,
                            float* __restrict__ out) {
    __shared__ float s[CC];
    int b = blockIdx.x, tid = threadIdx.x;
    if (tid < CC) s[tid] = 0.0f;
    __syncthreads();
    float p[CC];
    #pragma unroll
    for (int c = 0; c < CC; ++c) p[c] = 0.0f;
    for (int j = tid; j < K1; j += blockDim.x) {
        float tp = tanhf(pooled[(size_t)b * K1 + j]);
        const float* wr = Wh + (size_t)j * CC;
        #pragma unroll
        for (int c = 0; c < CC; ++c) p[c] += tp * wr[c];
    }
    #pragma unroll
    for (int c = 0; c < CC; ++c) atomicAdd(&s[c], p[c]);
    __syncthreads();
    if (tid < CC) out[b * CC + tid] = s[tid] + bh[tid];
}

// ---------------------------------------------------------------- launch
extern "C" void kernel_launch(void* const* d_in, const int* in_sizes, int n_in,
                              void* d_out, int out_size, void* d_ws, size_t ws_size,
                              hipStream_t stream) {
    const int*   x     = (const int*)  d_in[0];
    const float* embed = (const float*)d_in[1];
    const float* W0f   = (const float*)d_in[2];
    const float* b0f   = (const float*)d_in[3];
    const float* W0b   = (const float*)d_in[4];
    const float* b0b   = (const float*)d_in[5];
    const float* W1f   = (const float*)d_in[6];
    const float* b1f   = (const float*)d_in[7];
    const float* W1b   = (const float*)d_in[8];
    const float* b1b   = (const float*)d_in[9];
    const float* Wh    = (const float*)d_in[10];
    const float* bh    = (const float*)d_in[11];
    float* out = (float*)d_out;

    char* ws = (char*)d_ws;
    size_t off = 0;
    ushort_t* e_bf   = (ushort_t*)(ws + off); off += (size_t)TB * DP * 2;     // 20.97 MB
    ushort_t* h0_bf  = (ushort_t*)(ws + off); off += (size_t)TB * K1 * 2;     // 67.1 MB
    ushort_t* w0f_bf = (ushort_t*)(ws + off); off += (size_t)DP * NN * 2;     // [N,DP] transposed
    ushort_t* w0b_bf = (ushort_t*)(ws + off); off += (size_t)DP * NN * 2;
    ushort_t* w1f_bf = (ushort_t*)(ws + off); off += (size_t)K1 * NN * 2;     // [N,K1] transposed
    ushort_t* w1b_bf = (ushort_t*)(ws + off); off += (size_t)K1 * NN * 2;
    float*    Uf     = (float*)   (ws + off); off += (size_t)TB * NN * 4;     // 268 MB
    float*    Ub     = (float*)   (ws + off); off += (size_t)TB * NN * 4;     // 268 MB
    float*    pooled = (float*)   (ws + off); off += (size_t)BB * K1 * 4;
    (void)in_sizes; (void)n_in; (void)out_size; (void)ws_size;

    // 1) embedding gather -> bf16 (K padded to 320)
    embed_gather_kernel<<<TB, 64, 0, stream>>>(x, embed, e_bf);

    // 2) weights -> bf16, transposed to [N, Kp] (K zero-padded)
    cvt_weight_t_kernel<<<(NN * DP + 255) / 256, 256, 0, stream>>>(W0f, w0f_bf, DD, DP, NN);
    cvt_weight_t_kernel<<<(NN * DP + 255) / 256, 256, 0, stream>>>(W0b, w0b_bf, DD, DP, NN);
    cvt_weight_t_kernel<<<(NN * K1 + 255) / 256, 256, 0, stream>>>(W1f, w1f_bf, K1, K1, NN);
    cvt_weight_t_kernel<<<(NN * K1 + 255) / 256, 256, 0, stream>>>(W1b, w1b_bf, K1, K1, NN);

    dim3 ggrid(NN / NBLK, TB / MBLK);  // (16, 256)

    // 3) layer 0: both direction GEMMs, then fused bidirectional scan -> h0_bf
    wmma_gemm_bf16_kernel<<<ggrid, 256, 0, stream>>>(e_bf, w0f_bf, Uf, TB, NN, DP);
    wmma_gemm_bf16_kernel<<<ggrid, 256, 0, stream>>>(e_bf, w0b_bf, Ub, TB, NN, DP);
    sru_scan_bidir_kernel<<<(2 * BB * HH) / 256, 256, 0, stream>>>(
        Uf, Ub, b0f, b0b, h0_bf, (float*)nullptr);

    // 4) layer 1: GEMMs on h0, fused scan + tanh/max-pool -> pooled
    wmma_gemm_bf16_kernel<<<ggrid, 256, 0, stream>>>(h0_bf, w1f_bf, Uf, TB, NN, K1);
    wmma_gemm_bf16_kernel<<<ggrid, 256, 0, stream>>>(h0_bf, w1b_bf, Ub, TB, NN, K1);
    sru_scan_bidir_kernel<<<(2 * BB * HH) / 256, 256, 0, stream>>>(
        Uf, Ub, b1f, b1b, (ushort_t*)nullptr, pooled);

    // 5) head: tanh(pooled) @ Wh + bh -> [64,10]
    head_kernel<<<BB, 128, 0, stream>>>(pooled, Wh, bh, out);
}
